// MultiheadSelfAttention_62199716381475
// MI455X (gfx1250) — matrix operand order repaired and checked by reading, not compiled, not run on hardware
//
#include <hip/hip_runtime.h>

typedef __attribute__((ext_vector_type(16))) _Float16 v16h;
typedef __attribute__((ext_vector_type(8)))  _Float16 v8h;
typedef __attribute__((ext_vector_type(4)))  _Float16 v4h;
typedef __attribute__((ext_vector_type(8)))  float    v8f;

#define DM    768
#define HEADS 12
#define DH    64
#define SEQ   2048
#define BATCH 4
#define MTOT  (BATCH * SEQ)   // 8192 rows
#define NTILES (DM / 16)      // 48
#define BHTOT (BATCH * HEADS) // 48
#define QTILES (SEQ / 16)     // 128

static __device__ __forceinline__ v16h hcat(v8h lo, v8h hi) {
  return __builtin_shufflevector(lo, hi, 0,1,2,3,4,5,6,7,8,9,10,11,12,13,14,15);
}
static __device__ __forceinline__ v8h ld8(const _Float16* p) {
  return *reinterpret_cast<const v8h*>(p);
}
// A-fragment (16x32 f16): lane m=l15; halves j -> k=hi*16+j (16 contiguous halves)
static __device__ __forceinline__ v16h ldA(const _Float16* rowp, int k0) {
  return hcat(ld8(rowp + k0), ld8(rowp + k0 + 8));
}
// B-fragment (32x16 f16): lane n=l15; halves j -> k=hi*16+j (16 contiguous halves)
static __device__ __forceinline__ v16h ldB(const _Float16* rowp, int k0) {
  return hcat(ld8(rowp + k0), ld8(rowp + k0 + 8));
}
static __device__ __forceinline__ v8f wmma16(v16h a, v16h b, v8f c) {
  return __builtin_amdgcn_wmma_f32_16x16x32_f16(false, a, false, b, (short)0, c,
                                                false, false);
}

// ---------------------------------------------------------------- convert
__global__ void f32_to_f16_kernel(const float* __restrict__ s,
                                  _Float16* __restrict__ d, int n) {
  int i = (blockIdx.x * blockDim.x + threadIdx.x) * 4;
  if (i + 4 <= n) {
    float4 f = *reinterpret_cast<const float4*>(s + i);
    v4h h = { (_Float16)f.x, (_Float16)f.y, (_Float16)f.z, (_Float16)f.w };
    *reinterpret_cast<v4h*>(d + i) = h;
  } else {
    for (; i < n; ++i) d[i] = (_Float16)s[i];
  }
}

// ---------------------------------------------------------------- fused QKV
// One wave -> 32 rows x 16 cols of Q,K,V simultaneously.
// 2 A-fragments reused against 3 weight B-fragments: 6 WMMAs / 10 b128 loads
// per K-step, double-buffered. waves_per_eu(2) gives the wave enough VGPRs
// to keep both buffers + 6 accumulators resident (no scratch spills).
__global__ void __launch_bounds__(256)
__attribute__((amdgpu_waves_per_eu(2)))
qkv_proj_kernel(const _Float16* __restrict__ xh,
                const _Float16* __restrict__ wq,
                const _Float16* __restrict__ wk,
                const _Float16* __restrict__ wv,
                _Float16* __restrict__ Qh,   // (bh, s, 64)
                _Float16* __restrict__ Kh,   // (bh, s, 64)
                _Float16* __restrict__ Vt) { // (bh, 64, s)
  const int wave = blockIdx.x * (blockDim.x >> 5) + (threadIdx.x >> 5);
  const int lane = threadIdx.x & 31;
  const int hi = lane >> 4, l15 = lane & 15;
  const int mtile = wave / NTILES, ntile = wave % NTILES; // 256 x 48
  const int m0 = mtile * 32, n0 = ntile * 16;

  const _Float16* ar = xh + (size_t)(m0 + l15) * DM + hi * 16;
  const _Float16* bq = wq + (size_t)(n0 + l15) * DM + hi * 16;
  const _Float16* bk = wk + (size_t)(n0 + l15) * DM + hi * 16;
  const _Float16* bv = wv + (size_t)(n0 + l15) * DM + hi * 16;

  v8f acc[3][2];
  #pragma unroll
  for (int t = 0; t < 3; ++t) { acc[t][0] = v8f{}; acc[t][1] = v8f{}; }

  v16h ca0 = ldA(ar, 0), ca1 = ldA(ar + 16 * DM, 0);
  v16h cq = ldB(bq, 0), ck = ldB(bk, 0), cv = ldB(bv, 0);

  for (int k0 = 0; k0 < DM - 32; k0 += 32) {
    const int kn = k0 + 32;
    __builtin_prefetch(ar + kn + 32, 0, 1);
    v16h na0 = ldA(ar, kn), na1 = ldA(ar + 16 * DM, kn);
    v16h nq = ldB(bq, kn), nk = ldB(bk, kn), nv = ldB(bv, kn);

    acc[0][0] = wmma16(ca0, cq, acc[0][0]);
    acc[0][1] = wmma16(ca1, cq, acc[0][1]);
    acc[1][0] = wmma16(ca0, ck, acc[1][0]);
    acc[1][1] = wmma16(ca1, ck, acc[1][1]);
    acc[2][0] = wmma16(ca0, cv, acc[2][0]);
    acc[2][1] = wmma16(ca1, cv, acc[2][1]);

    ca0 = na0; ca1 = na1; cq = nq; ck = nk; cv = nv;
  }
  acc[0][0] = wmma16(ca0, cq, acc[0][0]);
  acc[0][1] = wmma16(ca1, cq, acc[0][1]);
  acc[1][0] = wmma16(ca0, ck, acc[1][0]);
  acc[1][1] = wmma16(ca1, ck, acc[1][1]);
  acc[2][0] = wmma16(ca0, cv, acc[2][0]);
  acc[2][1] = wmma16(ca1, cv, acc[2][1]);

  const int b  = m0 / SEQ;              // 32-row tile never straddles a batch
  const int nC = n0 + l15;
  const int h  = nC / DH, d = nC % DH;
  const int bh = b * HEADS + h;
  #pragma unroll
  for (int mi = 0; mi < 2; ++mi) {
    #pragma unroll
    for (int vv = 0; vv < 8; ++vv) {
      const int s = (m0 % SEQ) + mi * 16 + vv + 8 * hi;
      Qh[((size_t)bh * SEQ + s) * DH + d] = (_Float16)(acc[0][mi][vv] * 0.125f);
      Kh[((size_t)bh * SEQ + s) * DH + d] = (_Float16)acc[1][mi][vv];
      Vt[((size_t)bh * DH + d) * SEQ + s] = (_Float16)acc[2][mi][vv];
    }
  }
}

// ---------------------------------------------------------------- attention
// One wave per (batch-head, 16-query tile). Flash-style online softmax,
// keys processed 32 at a time (two 16-key score tiles -> one K=32 PV step).
__global__ void __launch_bounds__(256)
__attribute__((amdgpu_waves_per_eu(2)))
attn_kernel(const _Float16* __restrict__ Qh,
            const _Float16* __restrict__ Kh,
            const _Float16* __restrict__ Vt,
            _Float16* __restrict__ Ah) { // (b, s, 768)
  __shared__ _Float16 plds[8 * 16 * 40]; // per-wave 16x32 P tile, stride 40
  const int wave = blockIdx.x * (blockDim.x >> 5) + (threadIdx.x >> 5);
  const int lane = threadIdx.x & 31;
  const int hi = lane >> 4, l15 = lane & 15;
  const int bh = wave >> 7, qt = wave & (QTILES - 1);
  const int qBase = qt * 16;
  _Float16* pbuf = plds + (threadIdx.x >> 5) * 640;

  const _Float16* qrowp = Qh + ((size_t)bh * SEQ + qBase + l15) * DH;
  const v16h qa0 = hcat(ld8(qrowp + hi * 16),      ld8(qrowp + 8 + hi * 16));
  const v16h qa1 = hcat(ld8(qrowp + 32 + hi * 16), ld8(qrowp + 40 + hi * 16));

  v8f o[4] = {v8f{}, v8f{}, v8f{}, v8f{}};
  float mr[8], lr[8];
  #pragma unroll
  for (int i = 0; i < 8; ++i) { mr[i] = -1e30f; lr[i] = 0.0f; }

  const int nPairs = (qBase + 15) / 32 + 1;
  for (int p = 0; p < nPairs; ++p) {
    const int kb = p * 32;
    const _Float16* k0p = Kh + ((size_t)bh * SEQ + kb + l15) * DH + hi * 16;
    const _Float16* k1p = k0p + 16 * DH;
    v16h kb00 = hcat(ld8(k0p),      ld8(k0p + 8));
    v16h kb01 = hcat(ld8(k0p + 32), ld8(k0p + 40));
    v16h kb10 = hcat(ld8(k1p),      ld8(k1p + 8));
    v16h kb11 = hcat(ld8(k1p + 32), ld8(k1p + 40));

    v8f s0 = {}, s1 = {};
    s0 = wmma16(qa0, kb00, s0); s0 = wmma16(qa1, kb01, s0);
    s1 = wmma16(qa0, kb10, s1); s1 = wmma16(qa1, kb11, s1);

    const int col0 = kb + l15, col1 = col0 + 16;
    #pragma unroll
    for (int vv = 0; vv < 8; ++vv) {
      const int qrow = qBase + vv + 8 * hi;
      float a0 = (col0 <= qrow) ? s0[vv] : -1e30f;   // causal mask
      float a1 = (col1 <= qrow) ? s1[vv] : -1e30f;
      float mt = fmaxf(a0, a1);
      mt = fmaxf(mt, __shfl_xor(mt, 1, 32));
      mt = fmaxf(mt, __shfl_xor(mt, 2, 32));
      mt = fmaxf(mt, __shfl_xor(mt, 4, 32));
      mt = fmaxf(mt, __shfl_xor(mt, 8, 32));
      const float mn = fmaxf(mr[vv], mt);
      const float sc = __expf(mr[vv] - mn);
      const float p0 = __expf(a0 - mn);
      const float p1 = __expf(a1 - mn);
      float rs = p0 + p1;
      rs += __shfl_xor(rs, 1, 32);
      rs += __shfl_xor(rs, 2, 32);
      rs += __shfl_xor(rs, 4, 32);
      rs += __shfl_xor(rs, 8, 32);
      lr[vv] = lr[vv] * sc + rs;
      mr[vv] = mn;
      o[0][vv] *= sc; o[1][vv] *= sc; o[2][vv] *= sc; o[3][vv] *= sc;
      pbuf[(vv + 8 * hi) * 40 + l15]      = (_Float16)p0;
      pbuf[(vv + 8 * hi) * 40 + 16 + l15] = (_Float16)p1;
    }

    // reload P as an A-fragment; DS ops are in-order within a wave
    const _Float16* prow = pbuf + l15 * 40;
    v16h pa = hcat(*reinterpret_cast<const v8h*>(prow + hi * 16),
                   *reinterpret_cast<const v8h*>(prow + 8 + hi * 16));

    #pragma unroll
    for (int nt = 0; nt < 4; ++nt) {
      const _Float16* vr =
          Vt + ((size_t)bh * DH + nt * 16 + l15) * SEQ + kb + hi * 16;
      v16h vb = hcat(ld8(vr), ld8(vr + 8));
      o[nt] = wmma16(pa, vb, o[nt]);
    }
  }

  const int bb = bh / HEADS, hh = bh % HEADS;
  #pragma unroll
  for (int vv = 0; vv < 8; ++vv) {
    const float rl = 1.0f / lr[vv];
    const int s = qBase + vv + 8 * hi;
    const size_t rowoff = ((size_t)bb * SEQ + s) * DM + hh * DH + l15;
    #pragma unroll
    for (int nt = 0; nt < 4; ++nt)
      Ah[rowoff + nt * 16] = (_Float16)(o[nt][vv] * rl);
  }
}

// ---------------------------------------------------------------- out proj
// One wave -> 32 rows x 48 cols: 6 WMMAs / 10 b128 loads per K-step,
// double-buffered; waves_per_eu(2) keeps everything in registers.
__global__ void __launch_bounds__(256)
__attribute__((amdgpu_waves_per_eu(2)))
out_proj_kernel(const _Float16* __restrict__ ah,
                const _Float16* __restrict__ wo,
                float* __restrict__ out) {
  const int wave = blockIdx.x * (blockDim.x >> 5) + (threadIdx.x >> 5);
  const int lane = threadIdx.x & 31;
  const int hi = lane >> 4, l15 = lane & 15;
  const int mtile = wave / 16, ntile = wave % 16; // 256 x 16
  const int m0 = mtile * 32, n0 = ntile * 48;

  const _Float16* ar = ah + (size_t)(m0 + l15) * DM + hi * 16;
  const _Float16* b0 = wo + (size_t)(n0 + l15) * DM + hi * 16;
  const _Float16* b1 = b0 + (size_t)16 * DM;
  const _Float16* b2 = b0 + (size_t)32 * DM;

  v8f acc[2][3];
  #pragma unroll
  for (int mi = 0; mi < 2; ++mi)
    #pragma unroll
    for (int ni = 0; ni < 3; ++ni) acc[mi][ni] = v8f{};

  v16h ca0 = ldA(ar, 0), ca1 = ldA(ar + 16 * DM, 0);
  v16h cb0 = ldB(b0, 0), cb1 = ldB(b1, 0), cb2 = ldB(b2, 0);

  for (int k0 = 0; k0 < DM - 32; k0 += 32) {
    const int kn = k0 + 32;
    __builtin_prefetch(ar + kn + 32, 0, 1);
    v16h na0 = ldA(ar, kn), na1 = ldA(ar + 16 * DM, kn);
    v16h nb0 = ldB(b0, kn), nb1 = ldB(b1, kn), nb2 = ldB(b2, kn);

    acc[0][0] = wmma16(ca0, cb0, acc[0][0]);
    acc[1][0] = wmma16(ca1, cb0, acc[1][0]);
    acc[0][1] = wmma16(ca0, cb1, acc[0][1]);
    acc[1][1] = wmma16(ca1, cb1, acc[1][1]);
    acc[0][2] = wmma16(ca0, cb2, acc[0][2]);
    acc[1][2] = wmma16(ca1, cb2, acc[1][2]);

    ca0 = na0; ca1 = na1; cb0 = nb0; cb1 = nb1; cb2 = nb2;
  }
  acc[0][0] = wmma16(ca0, cb0, acc[0][0]);
  acc[1][0] = wmma16(ca1, cb0, acc[1][0]);
  acc[0][1] = wmma16(ca0, cb1, acc[0][1]);
  acc[1][1] = wmma16(ca1, cb1, acc[1][1]);
  acc[0][2] = wmma16(ca0, cb2, acc[0][2]);
  acc[1][2] = wmma16(ca1, cb2, acc[1][2]);

  #pragma unroll
  for (int mi = 0; mi < 2; ++mi)
    #pragma unroll
    for (int ni = 0; ni < 3; ++ni)
      #pragma unroll
      for (int vv = 0; vv < 8; ++vv)
        out[(size_t)(m0 + mi * 16 + vv + 8 * hi) * DM + n0 + ni * 16 + l15] =
            acc[mi][ni][vv];
}

// ---------------------------------------------------------------- launch
extern "C" void kernel_launch(void* const* d_in, const int* in_sizes, int n_in,
                              void* d_out, int out_size, void* d_ws,
                              size_t ws_size, hipStream_t stream) {
  (void)in_sizes; (void)n_in; (void)out_size; (void)ws_size;
  const float* x  = (const float*)d_in[0];
  const float* Wq = (const float*)d_in[1];
  const float* Wk = (const float*)d_in[2];
  const float* Wv = (const float*)d_in[3];
  const float* Wo = (const float*)d_in[4];
  float* out = (float*)d_out;

  const size_t NX = (size_t)MTOT * DM;  // 6,291,456
  const size_t NW = (size_t)DM * DM;    //   589,824
  _Float16* ws  = (_Float16*)d_ws;
  _Float16* xh  = ws;
  _Float16* wqh = xh  + NX;
  _Float16* wkh = wqh + NW;
  _Float16* wvh = wkh + NW;
  _Float16* woh = wvh + NW;
  _Float16* Qh  = woh + NW;
  _Float16* Kh  = Qh  + NX;
  _Float16* Vt  = Kh  + NX;
  _Float16* Ah  = Vt  + NX;

  f32_to_f16_kernel<<<(int)((NX / 4 + 255) / 256), 256, 0, stream>>>(x,  xh,  (int)NX);
  f32_to_f16_kernel<<<(int)((NW / 4 + 255) / 256), 256, 0, stream>>>(Wq, wqh, (int)NW);
  f32_to_f16_kernel<<<(int)((NW / 4 + 255) / 256), 256, 0, stream>>>(Wk, wkh, (int)NW);
  f32_to_f16_kernel<<<(int)((NW / 4 + 255) / 256), 256, 0, stream>>>(Wv, wvh, (int)NW);
  f32_to_f16_kernel<<<(int)((NW / 4 + 255) / 256), 256, 0, stream>>>(Wo, woh, (int)NW);

  // QKV: 256 * 48 = 12288 waves, 8 waves per block
  qkv_proj_kernel<<<256 * NTILES / 8, 256, 0, stream>>>(xh, wqh, wkh, wvh,
                                                        Qh, Kh, Vt);
  // attention: 48 * 128 = 6144 waves
  attn_kernel<<<BHTOT * QTILES / 8, 256, 0, stream>>>(Qh, Kh, Vt, Ah);

  // out projection: 256 * 16 = 4096 waves
  out_proj_kernel<<<256 * 16 / 8, 256, 0, stream>>>(Ah, woh, out);
}
